// AttentiveFP_65979287601501
// MI455X (gfx1250) — matrix operand order repaired and checked by reading, not compiled, not run on hardware
//
#include <hip/hip_runtime.h>
#include <hip/hip_bf16.h>
#include <cstdint>
#include <cstddef>

// ---------------------------------------------------------------------------
// AttentiveFP forward for MI455X (gfx1250, wave32, WMMA).
// Dense GEMMs -> v_wmma_f32_16x16x32_f16, 128x64 block tile (8 waves x 4 acc),
// B panel staged to LDS from a k-pair-packed global layout with
// GLOBAL_LOAD_ASYNC_TO_LDS_B128 when available (sync b128 fallback).
// Sparse/gather parts -> coalesced elementwise kernels + f32 atomics.
// ---------------------------------------------------------------------------

#define N_NODES 30000
#define N_EDGES 60000
#define N_GRAPH 1200
#define EMB 768
#define HID 384
#define OUTD 768
#define NEG_SLOPE 0.01f

#ifndef __has_builtin
#define __has_builtin(x) 0
#endif
#if __has_builtin(__builtin_amdgcn_global_load_async_to_lds_b128) && \
    __has_builtin(__builtin_amdgcn_s_wait_asynccnt)
#define USE_ASYNC_LDS 1
#else
#define USE_ASYNC_LDS 0
#endif

typedef __attribute__((ext_vector_type(16))) _Float16 v16h;
typedef __attribute__((ext_vector_type(8)))  _Float16 v8h;
typedef __attribute__((ext_vector_type(8)))  float    v8f;
typedef __attribute__((ext_vector_type(4)))  int      v4i;

__device__ __forceinline__ float lrelu(float v) { return v > 0.f ? v : NEG_SLOPE * v; }
__device__ __forceinline__ float actf(float v, int act) {
  if (act == 1) return lrelu(v);
  if (act == 2) return v > 0.f ? v : 0.f;
  return v;
}
// Order-preserving float<->uint encoding for atomicMax-based segment max.
__device__ __forceinline__ unsigned fenc(float f) {
  unsigned u = __float_as_uint(f);
  return (u & 0x80000000u) ? ~u : (u | 0x80000000u);
}
__device__ __forceinline__ float fdec(unsigned u) {
  u = (u & 0x80000000u) ? (u & 0x7FFFFFFFu) : ~u;
  return __uint_as_float(u);
}

// ---------------------------------------------------------------------------
// Pack f32 weight [K,Nn] -> f16 k-pair interleaved layout:
//   Bp[(k>>1)*2*Nn + c*2 + (k&1)] = W[k*Nn + c]
// A k-pair's 64-column slice is then 256 contiguous bytes in global memory,
// matching both the LDS fragment layout and 16-byte async-copy granularity.
// ---------------------------------------------------------------------------
__global__ void k_pack_b(const float* __restrict__ W, _Float16* __restrict__ out,
                         int K, int Nn) {
  int t = blockIdx.x * 256 + threadIdx.x;
  if (t >= K * Nn) return;
  int k = t / Nn, c = t - k * Nn;
  out[(size_t)(k >> 1) * 2 * Nn + c * 2 + (k & 1)] = (_Float16)W[t];
}

// ---------------------------------------------------------------------------
// WMMA GEMM:  C[M,Nn] = act( A_f16[M,K] @ B[K,Nn] + bias + Cadd )
// A: row-major f16. Bp: k-pair packed f16 (see k_pack_b).
// Block: 256 threads = 8 waves; block tile 128(M) x 64(N); each wave owns a
// 16x64 strip = 4 accumulators, so each A fragment (two b128 loads) feeds 4
// v_wmma. B panel (only K x 32: 8 x 16B per k-pair) staged once per block into dynamic LDS.
// All M multiples of 16, Nn multiples of 64, K multiples of 32 here.
// ---------------------------------------------------------------------------
__global__ __launch_bounds__(256)
void k_wmma_gemm(const _Float16* __restrict__ A, const _Float16* __restrict__ Bp,
                 const float* __restrict__ bias, const float* __restrict__ Cadd,
                 float* __restrict__ Cf, _Float16* __restrict__ Ch,
                 int M, int Nn, int K, int act)
{
  extern __shared__ _Float16 Bs[];   // (K/2) * 128 halves = K*128 bytes (<=144KB)
  const int n0 = blockIdx.x * 64;
  const int kp2 = K >> 1;            // number of k-pairs

  // Stage B panel: one 16-byte segment per (k-pair, seg) -> contiguous LDS.
  for (int t = threadIdx.x; t < kp2 * 8; t += 256) {
    int kp = t >> 3, seg = t & 7;
    const _Float16* g = Bp + (size_t)kp * 2 * Nn + (size_t)n0 * 2 + seg * 8;
    _Float16* l = Bs + kp * 128 + seg * 8;
#if USE_ASYNC_LDS
    __builtin_amdgcn_global_load_async_to_lds_b128(
        (__attribute__((address_space(1))) v4i*)g,
        (__attribute__((address_space(3))) v4i*)l, 0, 0);
#else
    *(v8h*)l = *(const v8h*)g;
#endif
  }
#if USE_ASYNC_LDS
  __builtin_amdgcn_s_wait_asynccnt(0);
#endif
  __syncthreads();

  const int wave = threadIdx.x >> 5;
  const int lane = threadIdx.x & 31;
  const int tileM = blockIdx.y * 8 + wave;
  if (tileM * 16 >= M) return;

  const int row  = lane & 15;    // A row within tile
  const int hsel = lane >> 4;    // half-wave select
  const int kA0  = hsel * 16;    // A: vgprs 0..7 hold k=kA0..kA0+15 (contiguous)
  const int kB0  = hsel * 16;    // B: k base for this half-wave
  const int col  = lane & 15;    // B/C/D column within 16-tile

  const _Float16* Arow = A + (size_t)(tileM * 16 + row) * K;
  v8f acc[4] = {v8f{}, v8f{}, v8f{}, v8f{}};

  for (int k0 = 0; k0 < K; k0 += 32) {
    v8h lo = *(const v8h*)(Arow + k0 + kA0);        // global_load_b128
    v8h hi = *(const v8h*)(Arow + k0 + kA0 + 8);    // global_load_b128
    v16h a;
#pragma unroll
    for (int i = 0; i < 8; ++i) { a[i] = lo[i]; a[8 + i] = hi[i]; }
    // element (k, c) lives at Bs[(k>>1)*128 + c*2 + (k&1)], c in [0,64)
    const _Float16* bbase = Bs + ((k0 + kB0) >> 1) * 128 + col * 2;
#pragma unroll
    for (int j = 0; j < 4; ++j) {
      v16h b;
#pragma unroll
      for (int i = 0; i < 8; ++i) {
        const _Float16* q = bbase + j * 32 + i * 128;
        b[2 * i]     = q[0];
        b[2 * i + 1] = q[1];
      }
      acc[j] = __builtin_amdgcn_wmma_f32_16x16x32_f16(
          false, a, false, b, (short)0, acc[j], false, false);
    }
  }

#pragma unroll
  for (int j = 0; j < 4; ++j) {
    int n = n0 + j * 16 + col;
    float bv = bias ? bias[n] : 0.f;
#pragma unroll
    for (int i = 0; i < 8; ++i) {
      int m = tileM * 16 + hsel * 8 + i;  // C/D: lanes 0-15 rows 0-7, 16-31 rows 8-15
      size_t off = (size_t)m * Nn + n;
      float v = acc[j][i] + bv;
      if (Cadd) v += Cadd[off];
      v = actf(v, act);
      if (Cf) Cf[off] = v;
      if (Ch) Ch[off] = (_Float16)v;
    }
  }
}

// ---------------------------------------------------------------------------
// Elementwise / gather / scatter kernels
// ---------------------------------------------------------------------------
// out[r,c] = sum_j emb[idx[r,j], c]   (embedding-bag sum), written as f16
__global__ void k_embed_sum(const int* __restrict__ idx, const float* __restrict__ emb,
                            _Float16* __restrict__ out, int rows, int nidx, int C) {
  int t = blockIdx.x * 256 + threadIdx.x;
  if (t >= rows * C) return;
  int r = t / C, c = t - r * C;
  float s = 0.f;
  for (int j = 0; j < nidx; ++j) s += emb[(size_t)idx[r * nidx + j] * C + c];
  out[t] = (_Float16)s;
}

__global__ void k_gather_rows_f16(const _Float16* __restrict__ X, const int* __restrict__ idx,
                                  _Float16* __restrict__ Y, int R, int C) {
  int t = blockIdx.x * 256 + threadIdx.x;
  if (t >= R * C) return;
  int r = t / C, c = t - r * C;
  Y[t] = X[(size_t)idx[r] * C + c];
}

// Per-row dot products (f32 rows): o1 = X.v1, o2 = X.v2 (v2 optional). 1 wave/row.
__global__ void k_rowdot2_f32(const float* __restrict__ X, const float* __restrict__ v1,
                              const float* __restrict__ v2, float* __restrict__ o1,
                              float* __restrict__ o2, int R, int C) {
  int w = (blockIdx.x * blockDim.x + threadIdx.x) >> 5;
  int lane = threadIdx.x & 31;
  if (w >= R) return;
  const float* row = X + (size_t)w * C;
  float s1 = 0.f, s2 = 0.f;
  for (int c = lane; c < C; c += 32) {
    float xv = row[c];
    s1 += xv * v1[c];
    if (v2) s2 += xv * v2[c];
  }
#pragma unroll
  for (int off = 16; off; off >>= 1) {
    s1 += __shfl_xor(s1, off, 32);
    if (v2) s2 += __shfl_xor(s2, off, 32);
  }
  if (lane == 0) { o1[w] = s1; if (v2) o2[w] = s2; }
}

__global__ void k_rowdot1_f16(const _Float16* __restrict__ X, const float* __restrict__ v1,
                              float* __restrict__ o1, int R, int C) {
  int w = (blockIdx.x * blockDim.x + threadIdx.x) >> 5;
  int lane = threadIdx.x & 31;
  if (w >= R) return;
  const _Float16* row = X + (size_t)w * C;
  float s1 = 0.f;
  for (int c = lane; c < C; c += 32) s1 += (float)row[c] * v1[c];
#pragma unroll
  for (int off = 16; off; off >>= 1) s1 += __shfl_xor(s1, off, 32);
  if (lane == 0) o1[w] = s1;
}

// a[e] = leaky(ml[e] + xr[dst[e]])      (GATEConv score)
__global__ void k_score_gate(const float* __restrict__ ml, const float* __restrict__ xr,
                             const int* __restrict__ dst, float* __restrict__ a, int n) {
  int e = blockIdx.x * 256 + threadIdx.x;
  if (e < n) a[e] = lrelu(ml[e] + xr[dst[e]]);
}
// a[e] = leaky(s[src[e]] + d[dst[e]])   (GATConv score)
__global__ void k_score2(const float* __restrict__ s, const float* __restrict__ d,
                         const int* __restrict__ src, const int* __restrict__ dst,
                         float* __restrict__ a, int n) {
  int e = blockIdx.x * 256 + threadIdx.x;
  if (e < n) a[e] = lrelu(s[src[e]] + d[dst[e]]);
}
// a[v] = leaky(anode[v] + ag[batch[v]]) (readout score)
__global__ void k_score_node(const float* __restrict__ anode, const float* __restrict__ ag,
                             const int* __restrict__ batch, float* __restrict__ a, int n) {
  int v = blockIdx.x * 256 + threadIdx.x;
  if (v < n) a[v] = lrelu(anode[v] + ag[batch[v]]);
}

// Segment softmax machinery.
__global__ void k_seg_init(unsigned* __restrict__ mx, float* __restrict__ sm, int S) {
  int s = blockIdx.x * 256 + threadIdx.x;
  if (s < S) { mx[s] = fenc(-__builtin_inff()); sm[s] = 0.f; }
}
__global__ void k_seg_amax(const float* __restrict__ a, const int* __restrict__ idx,
                           unsigned* __restrict__ mx, int n) {
  int i = blockIdx.x * 256 + threadIdx.x;
  if (i < n) atomicMax(&mx[idx[i]], fenc(a[i]));
}
__global__ void k_seg_expsum(const float* __restrict__ a, const int* __restrict__ idx,
                             const unsigned* __restrict__ mx, float* __restrict__ sm,
                             float* __restrict__ w, int n) {
  int i = blockIdx.x * 256 + threadIdx.x;
  if (i >= n) return;
  float ex = __expf(a[i] - fdec(mx[idx[i]]));
  w[i] = ex;
  atomicAdd(&sm[idx[i]], ex);
}
__global__ void k_seg_norm(float* __restrict__ w, const int* __restrict__ idx,
                           const float* __restrict__ sm, int n) {
  int i = blockIdx.x * 256 + threadIdx.x;
  if (i < n) w[i] = w[i] / (sm[idx[i]] + 1e-16f);
}

// In-place row scale of an f16 matrix by per-row weight.
__global__ void k_scale_rows_f16(_Float16* __restrict__ X, const float* __restrict__ w,
                                 int R, int C) {
  int t = blockIdx.x * 256 + threadIdx.x;
  if (t >= R * C) return;
  int r = t / C;
  X[t] = (_Float16)((float)X[t] * w[r]);
}

// H[r,c] = bias[c] (or 0)
__global__ void k_init_rows(float* __restrict__ H, const float* __restrict__ bias, int R, int C) {
  int t = blockIdx.x * 256 + threadIdx.x;
  if (t >= R * C) return;
  H[t] = bias ? bias[t % C] : 0.f;
}

// H[sidx[r], c] += (w ? w[r] : 1) * X[(gidx ? gidx[r] : r), c]
__global__ void k_scatter_wadd(const float* __restrict__ X, const int* __restrict__ gidx,
                               const int* __restrict__ sidx, const float* __restrict__ w,
                               float* __restrict__ H, int R, int C) {
  int t = blockIdx.x * 256 + threadIdx.x;
  if (t >= R * C) return;
  int r = t / C, c = t - r * C;
  int rin = gidx ? gidx[r] : r;
  float v = X[(size_t)rin * C + c];
  if (w) v *= w[r];
  atomicAdd(&H[(size_t)sidx[r] * C + c], v);
}

// y_f16[t] = elu(x[t])
__global__ void k_elu_f16(const float* __restrict__ x, _Float16* __restrict__ y, int n) {
  int t = blockIdx.x * 256 + threadIdx.x;
  if (t >= n) return;
  float v = x[t];
  y[t] = (_Float16)(v > 0.f ? v : __expf(v) - 1.f);
}

// in-place relu on f32, with f16 copy
__global__ void k_relu_cvt(float* __restrict__ x, _Float16* __restrict__ y, int n) {
  int t = blockIdx.x * 256 + threadIdx.x;
  if (t >= n) return;
  float v = x[t];
  v = v > 0.f ? v : 0.f;
  x[t] = v;
  y[t] = (_Float16)v;
}

// GRU gate combine + relu (all GRU outputs in this model are relu-wrapped).
__global__ void k_gru(const float* __restrict__ gi, const float* __restrict__ gh,
                      const float* __restrict__ hst, float* __restrict__ of,
                      _Float16* __restrict__ oh, int R, int C) {
  int t = blockIdx.x * 256 + threadIdx.x;
  if (t >= R * C) return;
  int r = t / C, c = t - r * C;
  size_t b = (size_t)r * 3 * C;
  float ir = gi[b + c], iz = gi[b + C + c], in = gi[b + 2 * C + c];
  float hr = gh[b + c], hz = gh[b + C + c], hn = gh[b + 2 * C + c];
  float rr = 1.f / (1.f + __expf(-(ir + hr)));
  float zz = 1.f / (1.f + __expf(-(iz + hz)));
  float nn = tanhf(in + rr * hn);
  float v = (1.f - zz) * nn + zz * hst[t];
  v = v > 0.f ? v : 0.f;
  of[t] = v;
  if (oh) oh[t] = (_Float16)v;
}

// ---------------------------------------------------------------------------
// Host-side orchestration
// ---------------------------------------------------------------------------
static inline unsigned b1d(long n) { return (unsigned)((n + 255) / 256); }

extern "C" void kernel_launch(void* const* d_in, const int* in_sizes, int n_in,
                              void* d_out, int out_size, void* d_ws, size_t ws_size,
                              hipStream_t stream) {
  (void)in_sizes; (void)n_in; (void)out_size; (void)ws_size;

  const int*   x_idx         = (const int*)d_in[0];
  const int*   edge_index    = (const int*)d_in[1];   // [2, E]
  const int*   edge_attr_idx = (const int*)d_in[2];
  const int*   batch         = (const int*)d_in[3];
  const float* x_emb   = (const float*)d_in[4];
  const float* e_emb   = (const float*)d_in[5];
  const float* lin1_W  = (const float*)d_in[6];
  const float* lin1_b  = (const float*)d_in[7];
  const float* att_l   = (const float*)d_in[8];
  const float* att_r   = (const float*)d_in[9];
  const float* gl1_W   = (const float*)d_in[10];
  const float* gl2_W   = (const float*)d_in[11];
  const float* gate_b  = (const float*)d_in[12];
  const float* gru1_Wi = (const float*)d_in[13];
  const float* gru1_Wh = (const float*)d_in[14];
  const float* gru1_bi = (const float*)d_in[15];
  const float* gru1_bh = (const float*)d_in[16];
  const float* gat_W   = (const float*)d_in[17];
  const float* gat_as  = (const float*)d_in[18];
  const float* gat_ad  = (const float*)d_in[19];
  const float* gat_b   = (const float*)d_in[20];
  const float* gru2_Wi = (const float*)d_in[21];
  const float* gru2_Wh = (const float*)d_in[22];
  const float* gru2_bi = (const float*)d_in[23];
  const float* gru2_bh = (const float*)d_in[24];
  const float* mol_W   = (const float*)d_in[25];
  const float* mol_as  = (const float*)d_in[26];
  const float* mol_ad  = (const float*)d_in[27];
  const float* mol_b   = (const float*)d_in[28];
  const float* mgru_Wi = (const float*)d_in[29];
  const float* mgru_Wh = (const float*)d_in[30];
  const float* mgru_bi = (const float*)d_in[31];
  const float* mgru_bh = (const float*)d_in[32];
  const float* lin2_W  = (const float*)d_in[33];
  const float* lin2_b  = (const float*)d_in[34];

  const int* src = edge_index;
  const int* dst = edge_index + N_EDGES;

  // ---- workspace bump allocator ----
  char* p = (char*)d_ws;
  auto alloc = [&](size_t bytes) -> char* {
    char* r = p;
    p += (bytes + 255) & ~(size_t)255;
    return r;
  };
  // f16 packed weights (k-pair interleaved)
  _Float16* lin1h  = (_Float16*)alloc((size_t)EMB * HID * 2);
  _Float16* gl1hA  = (_Float16*)alloc((size_t)HID * HID * 2);   // gl1_W rows [0,HID)
  _Float16* gl1hB  = (_Float16*)alloc((size_t)EMB * HID * 2);   // gl1_W rows [HID,HID+EMB)
  _Float16* gl2h   = (_Float16*)alloc((size_t)HID * HID * 2);
  _Float16* g1ih   = (_Float16*)alloc((size_t)HID * 3 * HID * 2);
  _Float16* g1hh   = (_Float16*)alloc((size_t)HID * 3 * HID * 2);
  _Float16* gath   = (_Float16*)alloc((size_t)HID * HID * 2);
  _Float16* g2ih   = (_Float16*)alloc((size_t)HID * 3 * HID * 2);
  _Float16* g2hh   = (_Float16*)alloc((size_t)HID * 3 * HID * 2);
  _Float16* molh   = (_Float16*)alloc((size_t)HID * HID * 2);
  _Float16* mgih   = (_Float16*)alloc((size_t)HID * 3 * HID * 2);
  _Float16* mghh   = (_Float16*)alloc((size_t)HID * 3 * HID * 2);
  _Float16* lin2h  = (_Float16*)alloc((size_t)HID * OUTD * 2);
  // activations
  _Float16* xnodeh = (_Float16*)alloc((size_t)N_NODES * EMB * 2);
  _Float16* eah    = (_Float16*)alloc((size_t)N_EDGES * EMB * 2);
  float*    xf     = (float*)alloc((size_t)N_NODES * HID * 4);   // node state f32
  _Float16* xh     = (_Float16*)alloc((size_t)N_NODES * HID * 2);
  float*    ebig   = (float*)alloc((size_t)N_EDGES * HID * 4);   // t_e, then p
  _Float16* srch   = (_Float16*)alloc((size_t)N_EDGES * HID * 2);
  _Float16* mh     = (_Float16*)alloc((size_t)N_EDGES * HID * 2);
  float*    hbuf   = (float*)alloc((size_t)N_NODES * HID * 4);
  _Float16* ehh    = (_Float16*)alloc((size_t)N_NODES * HID * 2);
  float*    gi     = (float*)alloc((size_t)N_NODES * 3 * HID * 4);
  float*    gh     = (float*)alloc((size_t)N_NODES * 3 * HID * 4);
  float*    xl     = (float*)alloc((size_t)N_NODES * HID * 4);
  float*    score  = (float*)alloc((size_t)N_EDGES * 4);
  float*    wgt    = (float*)alloc((size_t)N_EDGES * 4);
  float*    ml     = (float*)alloc((size_t)N_EDGES * 4);
  float*    xr     = (float*)alloc((size_t)N_NODES * 4);
  float*    snode  = (float*)alloc((size_t)N_NODES * 4);
  float*    dnode  = (float*)alloc((size_t)N_NODES * 4);
  float*    anode  = (float*)alloc((size_t)N_NODES * 4);
  float*    ag     = (float*)alloc((size_t)N_GRAPH * 4);
  unsigned* mx     = (unsigned*)alloc((size_t)N_NODES * 4);
  float*    sm     = (float*)alloc((size_t)N_NODES * 4);
  float*    outg   = (float*)alloc((size_t)N_GRAPH * HID * 4);
  _Float16* ogh    = (_Float16*)alloc((size_t)N_GRAPH * HID * 2);
  float*    hg     = (float*)alloc((size_t)N_GRAPH * HID * 4);
  float*    tg     = (float*)alloc((size_t)N_GRAPH * HID * 4);

  auto pack = [&](const float* W, _Float16* d, int K, int Nn) {
    k_pack_b<<<b1d((long)K * Nn), 256, 0, stream>>>(W, d, K, Nn);
  };
  auto gemm = [&](const _Float16* A, const _Float16* Bp, const float* bias, const float* Cadd,
                  float* Cf, _Float16* Ch, int M, int Nn, int K, int act) {
    dim3 grid(Nn / 64, (M / 16 + 7) / 8);
    size_t shmem = (size_t)K * 128;  // (K/2) k-pairs * 128 halves * 2B
    k_wmma_gemm<<<grid, 256, shmem, stream>>>(A, Bp, bias, Cadd, Cf, Ch, M, Nn, K, act);
  };
  auto seg_softmax = [&](float* a, float* w, const int* idx, int n, int S) {
    k_seg_init<<<b1d(S), 256, 0, stream>>>(mx, sm, S);
    k_seg_amax<<<b1d(n), 256, 0, stream>>>(a, idx, mx, n);
    k_seg_expsum<<<b1d(n), 256, 0, stream>>>(a, idx, mx, sm, w, n);
    k_seg_norm<<<b1d(n), 256, 0, stream>>>(w, idx, sm, n);
  };

  // ---- weight conversion + packing ----
  pack(lin1_W, lin1h, EMB, HID);
  pack(gl1_W, gl1hA, HID, HID);                     // rows [0, HID)
  pack(gl1_W + (size_t)HID * HID, gl1hB, EMB, HID); // rows [HID, HID+EMB)
  pack(gl2_W, gl2h, HID, HID);
  pack(gru1_Wi, g1ih, HID, 3 * HID);
  pack(gru1_Wh, g1hh, HID, 3 * HID);
  pack(gat_W, gath, HID, HID);
  pack(gru2_Wi, g2ih, HID, 3 * HID);
  pack(gru2_Wh, g2hh, HID, 3 * HID);
  pack(mol_W, molh, HID, HID);
  pack(mgru_Wi, mgih, HID, 3 * HID);
  pack(mgru_Wh, mghh, HID, 3 * HID);
  pack(lin2_W, lin2h, HID, OUTD);

  // ---- embeddings + lin1 ----
  k_embed_sum<<<b1d((long)N_NODES * EMB), 256, 0, stream>>>(x_idx, x_emb, xnodeh, N_NODES, 9, EMB);
  k_embed_sum<<<b1d((long)N_EDGES * EMB), 256, 0, stream>>>(edge_attr_idx, e_emb, eah, N_EDGES, 3, EMB);
  gemm(xnodeh, lin1h, lin1_b, nullptr, xf, xh, N_NODES, HID, EMB, /*leaky*/1);

  // ---- GATEConv ----
  gemm(eah, gl1hB, nullptr, nullptr, ebig, nullptr, N_EDGES, HID, EMB, 0);  // t_e
  k_gather_rows_f16<<<b1d((long)N_EDGES * HID), 256, 0, stream>>>(xh, src, srch, N_EDGES, HID);
  gemm(srch, gl1hA, nullptr, ebig, nullptr, mh, N_EDGES, HID, HID, 1);      // m
  k_rowdot1_f16<<<(N_EDGES + 7) / 8, 256, 0, stream>>>(mh, att_l, ml, N_EDGES, HID);
  k_rowdot2_f32<<<(N_NODES + 7) / 8, 256, 0, stream>>>(xf, att_r, nullptr, xr, nullptr, N_NODES, HID);
  k_score_gate<<<b1d(N_EDGES), 256, 0, stream>>>(ml, xr, dst, score, N_EDGES);
  seg_softmax(score, wgt, dst, N_EDGES, N_NODES);
  k_scale_rows_f16<<<b1d((long)N_EDGES * HID), 256, 0, stream>>>(mh, wgt, N_EDGES, HID);
  gemm(mh, gl2h, nullptr, nullptr, ebig, nullptr, N_EDGES, HID, HID, 0);    // (m*a)@gl2
  k_init_rows<<<b1d((long)N_NODES * HID), 256, 0, stream>>>(hbuf, gate_b, N_NODES, HID);
  k_scatter_wadd<<<b1d((long)N_EDGES * HID), 256, 0, stream>>>(ebig, nullptr, dst, nullptr, hbuf, N_EDGES, HID);
  k_elu_f16<<<b1d((long)N_NODES * HID), 256, 0, stream>>>(hbuf, ehh, N_NODES * HID);
  gemm(ehh, g1ih, gru1_bi, nullptr, gi, nullptr, N_NODES, 3 * HID, HID, 0);
  gemm(xh, g1hh, gru1_bh, nullptr, gh, nullptr, N_NODES, 3 * HID, HID, 0);
  k_gru<<<b1d((long)N_NODES * HID), 256, 0, stream>>>(gi, gh, xf, xf, xh, N_NODES, HID);

  // ---- atom GATConv ----
  gemm(xh, gath, nullptr, nullptr, xl, nullptr, N_NODES, HID, HID, 0);
  k_rowdot2_f32<<<(N_NODES + 7) / 8, 256, 0, stream>>>(xl, gat_as, gat_ad, snode, dnode, N_NODES, HID);
  k_score2<<<b1d(N_EDGES), 256, 0, stream>>>(snode, dnode, src, dst, score, N_EDGES);
  seg_softmax(score, wgt, dst, N_EDGES, N_NODES);
  k_init_rows<<<b1d((long)N_NODES * HID), 256, 0, stream>>>(hbuf, gat_b, N_NODES, HID);
  k_scatter_wadd<<<b1d((long)N_EDGES * HID), 256, 0, stream>>>(xl, src, dst, wgt, hbuf, N_EDGES, HID);
  k_elu_f16<<<b1d((long)N_NODES * HID), 256, 0, stream>>>(hbuf, ehh, N_NODES * HID);
  gemm(ehh, g2ih, gru2_bi, nullptr, gi, nullptr, N_NODES, 3 * HID, HID, 0);
  gemm(xh, g2hh, gru2_bh, nullptr, gh, nullptr, N_NODES, 3 * HID, HID, 0);
  k_gru<<<b1d((long)N_NODES * HID), 256, 0, stream>>>(gi, gh, xf, xf, xh, N_NODES, HID);

  // ---- attentive readout, 3 timesteps ----
  k_init_rows<<<b1d((long)N_GRAPH * HID), 256, 0, stream>>>(outg, nullptr, N_GRAPH, HID);
  k_scatter_wadd<<<b1d((long)N_NODES * HID), 256, 0, stream>>>(xf, nullptr, batch, nullptr, outg, N_NODES, HID);
  k_relu_cvt<<<b1d((long)N_GRAPH * HID), 256, 0, stream>>>(outg, ogh, N_GRAPH * HID);
  gemm(xh, molh, nullptr, nullptr, xl, nullptr, N_NODES, HID, HID, 0);   // xl = x @ mol_W
  k_rowdot2_f32<<<(N_NODES + 7) / 8, 256, 0, stream>>>(xl, mol_as, nullptr, anode, nullptr, N_NODES, HID);

  for (int ts = 0; ts < 3; ++ts) {
    gemm(ogh, molh, nullptr, nullptr, tg, nullptr, N_GRAPH, HID, HID, 0);  // out@mol_W
    k_rowdot2_f32<<<(N_GRAPH + 7) / 8, 256, 0, stream>>>(tg, mol_ad, nullptr, ag, nullptr, N_GRAPH, HID);
    k_score_node<<<b1d(N_NODES), 256, 0, stream>>>(anode, ag, batch, score, N_NODES);
    seg_softmax(score, wgt, batch, N_NODES, N_GRAPH);
    k_init_rows<<<b1d((long)N_GRAPH * HID), 256, 0, stream>>>(hg, mol_b, N_GRAPH, HID);
    k_scatter_wadd<<<b1d((long)N_NODES * HID), 256, 0, stream>>>(xl, nullptr, batch, wgt, hg, N_NODES, HID);
    k_elu_f16<<<b1d((long)N_GRAPH * HID), 256, 0, stream>>>(hg, ehh, N_GRAPH * HID);
    gemm(ehh, mgih, mgru_bi, nullptr, gi, nullptr, N_GRAPH, 3 * HID, HID, 0);
    gemm(ogh, mghh, mgru_bh, nullptr, gh, nullptr, N_GRAPH, 3 * HID, HID, 0);
    k_gru<<<b1d((long)N_GRAPH * HID), 256, 0, stream>>>(gi, gh, outg, outg, ogh, N_GRAPH, HID);
  }

  // ---- final projection -> d_out [G, OUT] f32 ----
  gemm(ogh, lin2h, lin2_b, nullptr, (float*)d_out, nullptr, N_GRAPH, OUTD, HID, 0);
}
